// CausalSelfAttention_21208548508482
// MI455X (gfx1250) — matrix operand order repaired and checked by reading, not compiled, not run on hardware
//
#include <hip/hip_runtime.h>

typedef __bf16 bf16;
typedef __attribute__((ext_vector_type(16))) __bf16 v16bf;
typedef __attribute__((ext_vector_type(8)))  __bf16 v8bf;
typedef __attribute__((ext_vector_type(8)))  float   v8f;

#define D_MODEL 1024
#define N_HEADS 16
#define D_HEAD  64
#define B_SZ    4
#define T_SEQ   2048
#define M_ROWS  (B_SZ * T_SEQ)   // 8192

static __device__ inline v16bf cat8(v8bf lo, v8bf hi) {
  return __builtin_shufflevector(lo, hi, 0,1,2,3,4,5,6,7,8,9,10,11,12,13,14,15);
}

static __device__ inline v16bf ldA(const bf16* p) {       // 16 bf16 as 2x16B
  return cat8(*(const v8bf*)p, *(const v8bf*)(p + 8));
}

static __device__ inline v8f wmma_bf16(v16bf a, v16bf b, v8f c) {
  return __builtin_amdgcn_wmma_f32_16x16x32_bf16(false, a, false, b, (short)0, c, false, false);
}

// ---------------- conversion / packing ----------------

__global__ void k_f32_to_bf16(const float* __restrict__ in, bf16* __restrict__ out, int n) {
  int i = blockIdx.x * blockDim.x + threadIdx.x;
  if (i < n) out[i] = (bf16)in[i];
}

// Pack row-major fp32 W[K,N] into B-fragment-major bf16:
// packed[((kt*NT + nt)*32 + lane)*16 + e] = W[kt*32 + (lane<16?0:16) + e][nt*16 + (lane&15)]
__global__ void k_pack_w(const float* __restrict__ w, bf16* __restrict__ wp,
                         int N, int NT, int total) {
  int i = blockIdx.x * blockDim.x + threadIdx.x;
  if (i >= total) return;
  int e    = i & 15;
  int lane = (i >> 4) & 31;
  int tile = i >> 9;
  int nt = tile % NT;
  int kt = tile / NT;
  int n = nt * 16 + (lane & 15);
  int k = kt * 32 + ((lane < 16) ? 0 : 16) + e;
  wp[i] = (bf16)w[(size_t)k * N + n];
}

// ---------------- wave-level 32x64 GEMM core, double-buffered ----------------
// A: row-major bf16 [M,K]; Bp: fragment-packed bf16. 8 WMMAs per 32-wide k-step.
__device__ inline void wave_gemm_32x64(const bf16* __restrict__ A, const bf16* __restrict__ Bp,
                                       int K, int NT, int mrow0, int ntile0, v8f acc[8]) {
  int lane = threadIdx.x & 31;
  int koff = (lane & 16) ? 16 : 0;          // A-frag K-offset: contiguous k = 16h + s
  const bf16* pa0 = A + (size_t)(mrow0 + (lane & 15)) * K + koff;
  const bf16* pa1 = pa0 + (size_t)16 * K;
  const bf16* pb  = Bp + ((size_t)ntile0 * 32 + lane) * 16;
  const size_t bstep = (size_t)NT * 512;    // packed elements per k-step

  v16bf ac0 = ldA(pa0), ac1 = ldA(pa1);
  v16bf bc[4];
#pragma unroll
  for (int c = 0; c < 4; ++c) bc[c] = *(const v16bf*)(pb + c * 512);

  int KT = K >> 5;
  for (int kt = 0; kt < KT - 1; ++kt) {
    pa0 += 32; pa1 += 32; pb += bstep;
    // prefetch next k-step fragments
    v16bf an0 = ldA(pa0), an1 = ldA(pa1);
    v16bf bn[4];
#pragma unroll
    for (int c = 0; c < 4; ++c) bn[c] = *(const v16bf*)(pb + c * 512);
    // consume current fragments
#pragma unroll
    for (int c = 0; c < 4; ++c) acc[c]     = wmma_bf16(ac0, bc[c], acc[c]);
#pragma unroll
    for (int c = 0; c < 4; ++c) acc[4 + c] = wmma_bf16(ac1, bc[c], acc[4 + c]);
    ac0 = an0; ac1 = an1;
#pragma unroll
    for (int c = 0; c < 4; ++c) bc[c] = bn[c];
  }
#pragma unroll
  for (int c = 0; c < 4; ++c) acc[c]     = wmma_bf16(ac0, bc[c], acc[c]);
#pragma unroll
  for (int c = 0; c < 4; ++c) acc[4 + c] = wmma_bf16(ac1, bc[c], acc[4 + c]);
}

// ---------------- stage 1: qkv = x @ Wqkv, scatter to Q,K,Vt ----------------

__global__ __launch_bounds__(256) void k_gemm_qkv(const bf16* __restrict__ xb, const bf16* __restrict__ wp,
                           bf16* __restrict__ Qb, bf16* __restrict__ Kb, bf16* __restrict__ Vt) {
  int wid  = blockIdx.x * (blockDim.x >> 5) + (threadIdx.x >> 5);
  int lane = threadIdx.x & 31;
  const int NT = (3 * D_MODEL) / 16;        // 192
  int mt  = wid / 48;                       // 256 m-tiles of 32 rows
  int nt4 = wid % 48;                       // 48 col groups of 64
  v8f acc[8] = {};
  wave_gemm_32x64(xb, wp, D_MODEL, NT, mt * 32, nt4 * 4, acc);

#pragma unroll
  for (int mi = 0; mi < 2; ++mi) {
    int rbase = mt * 32 + mi * 16 + ((lane & 16) ? 8 : 0);
#pragma unroll
    for (int c = 0; c < 4; ++c) {
      int col = nt4 * 64 + c * 16 + (lane & 15);
      int s = col >> 10;              // 0:q 1:k 2:v
      int h = (col & 1023) >> 6;
      int d = col & 63;
#pragma unroll
      for (int r = 0; r < 8; ++r) {
        int row = rbase + r;
        int b = row >> 11;
        int t = row & 2047;
        float v = acc[mi * 4 + c][r];
        size_t bh = (size_t)(b * N_HEADS + h);
        if (s == 0)      Qb[(bh * T_SEQ + t) * D_HEAD + d] = (bf16)(v * 0.125f); // 1/sqrt(64)
        else if (s == 1) Kb[(bh * T_SEQ + t) * D_HEAD + d] = (bf16)v;
        else             Vt[(bh * D_HEAD + d) * T_SEQ + t] = (bf16)v;
      }
    }
  }
}

// ---------------- stage 2: flash attention, one wave per 16-query tile ----------------

__global__ __launch_bounds__(256) void k_attention(const bf16* __restrict__ Qb, const bf16* __restrict__ Kb,
                            const bf16* __restrict__ Vt, bf16* __restrict__ Yb) {
  __shared__ bf16 plds[8 * 16 * 32];        // 1KB per wave P-staging
  int wib  = threadIdx.x >> 5;
  int lane = threadIdx.x & 31;
  int wid  = blockIdx.x * 8 + wib;
  int qt = wid & 127;                        // T/16 = 128 query tiles
  int bh = wid >> 7;                         // b*16+h, 0..63
  int qbase = qt * 16;

  const bf16* Qh = Qb + (size_t)bh * T_SEQ * D_HEAD;
  const bf16* Kh = Kb + (size_t)bh * T_SEQ * D_HEAD;
  const bf16* Vh = Vt + (size_t)bh * D_HEAD * T_SEQ;
  bf16* ldsw = plds + wib * 16 * 32;

  int hrow = lane & 15;
  int koff = (lane & 16) ? 16 : 0;
  int roff = (lane & 16) ? 16 : 0;

  // Q as two 16x32 A-fragments (Dh = 64)
  v16bf aq[2];
#pragma unroll
  for (int ks = 0; ks < 2; ++ks)
    aq[ks] = ldA(Qh + (size_t)(qbase + hrow) * D_HEAD + ks * 32 + koff);

  v8f o[4] = {};
  float mrow[8], lrow[8];
#pragma unroll
  for (int r = 0; r < 8; ++r) { mrow[r] = -1e30f; lrow[r] = 0.0f; }

  int kend = qbase + 16;
  for (int kb = 0; kb < kend; kb += 32) {
    // ---- S = Q K^T for a 16q x 32k block (two 16x16 C-frags) ----
    v8f s0 = {}, s1 = {};
    int keyA = kb + (lane & 15);
    int keyB = keyA + 16;
#pragma unroll
    for (int ks = 0; ks < 2; ++ks) {
      v16bf b0 = *(const v16bf*)(Kh + (size_t)keyA * D_HEAD + ks * 32 + roff);
      v16bf b1 = *(const v16bf*)(Kh + (size_t)keyB * D_HEAD + ks * 32 + roff);
      s0 = wmma_bf16(aq[ks], b0, s0);
      s1 = wmma_bf16(aq[ks], b1, s1);
    }
    // ---- hoist V B-fragments: independent of softmax, overlap with VALU ----
    v16bf bv[4];
#pragma unroll
    for (int c = 0; c < 4; ++c)
      bv[c] = *(const v16bf*)(Vh + (size_t)(c * 16 + (lane & 15)) * T_SEQ + kb + roff);
    // ---- causal mask + online softmax (rows = VGPR idx; keys across 16 lanes) ----
#pragma unroll
    for (int r = 0; r < 8; ++r) {
      int qi = qbase + ((lane & 16) ? 8 : 0) + r;
      float v0 = s0[r], v1 = s1[r];
      if (keyA > qi) v0 = -1e30f;
      if (keyB > qi) v1 = -1e30f;
      float tmax = fmaxf(v0, v1);
#pragma unroll
      for (int m = 8; m >= 1; m >>= 1)
        tmax = fmaxf(tmax, __shfl_xor(tmax, m, 16));
      float mnew = fmaxf(mrow[r], tmax);
      float csc  = __expf(mrow[r] - mnew);
      mrow[r] = mnew;
      float p0 = __expf(v0 - mnew);
      float p1 = __expf(v1 - mnew);
      float ps = p0 + p1;
#pragma unroll
      for (int m = 8; m >= 1; m >>= 1)
        ps += __shfl_xor(ps, m, 16);
      lrow[r] = lrow[r] * csc + ps;
#pragma unroll
      for (int c = 0; c < 4; ++c) o[c][r] *= csc;
      int prow = ((lane & 16) ? 8 : 0) + r;
      ldsw[prow * 32 + (lane & 15)]      = (bf16)p0;
      ldsw[prow * 32 + 16 + (lane & 15)] = (bf16)p1;
    }
    // ---- reload P as 16x32 A-fragment (wave-private LDS round trip) ----
    v16bf pa = ldA(ldsw + hrow * 32 + koff);
    // ---- O += P @ V ----
#pragma unroll
    for (int c = 0; c < 4; ++c)
      o[c] = wmma_bf16(pa, bv[c], o[c]);
  }

  // ---- normalize and write y (merge heads): Yb[B*T, D_MODEL] bf16 ----
  int b = bh >> 4, h = bh & 15;
#pragma unroll
  for (int c = 0; c < 4; ++c) {
    int d = c * 16 + (lane & 15);
#pragma unroll
    for (int r = 0; r < 8; ++r) {
      int row = ((lane & 16) ? 8 : 0) + r;
      float v = o[c][r] / lrow[r];
      int t = qbase + row;
      Yb[((size_t)(b * T_SEQ + t)) * D_MODEL + h * D_HEAD + d] = (bf16)v;
    }
  }
}

// ---------------- stage 3: out = y @ Wproj (fp32 out) ----------------

__global__ __launch_bounds__(256) void k_gemm_proj(const bf16* __restrict__ Yb, const bf16* __restrict__ wp,
                            float* __restrict__ out) {
  int wid  = blockIdx.x * (blockDim.x >> 5) + (threadIdx.x >> 5);
  int lane = threadIdx.x & 31;
  const int NT = D_MODEL / 16;              // 64
  int mt  = wid >> 4;                       // 256 m-tiles of 32 rows
  int nt4 = wid & 15;                       // 16 col groups of 64
  v8f acc[8] = {};
  wave_gemm_32x64(Yb, wp, D_MODEL, NT, mt * 32, nt4 * 4, acc);
#pragma unroll
  for (int mi = 0; mi < 2; ++mi) {
    int rbase = mt * 32 + mi * 16 + ((lane & 16) ? 8 : 0);
#pragma unroll
    for (int c = 0; c < 4; ++c) {
      int col = nt4 * 64 + c * 16 + (lane & 15);
#pragma unroll
      for (int r = 0; r < 8; ++r)
        out[(size_t)(rbase + r) * D_MODEL + col] = acc[mi * 4 + c][r];
    }
  }
}

// ---------------- host launch ----------------

extern "C" void kernel_launch(void* const* d_in, const int* in_sizes, int n_in,
                              void* d_out, int out_size, void* d_ws, size_t ws_size,
                              hipStream_t stream) {
  const float* x     = (const float*)d_in[0];
  const float* wqkv  = (const float*)d_in[1];
  const float* wproj = (const float*)d_in[2];
  float* out = (float*)d_out;

  char* ws = (char*)d_ws;
  size_t off = 0;
  auto carve = [&](size_t bytes) -> void* {
    void* p = ws + off;
    off += (bytes + 255) & ~(size_t)255;
    return p;
  };
  const size_t NE = (size_t)M_ROWS * D_MODEL;          // 8388608
  bf16* xb      = (bf16*)carve(NE * 2);
  bf16* wqkv_p  = (bf16*)carve((size_t)D_MODEL * 3 * D_MODEL * 2);
  bf16* wproj_p = (bf16*)carve((size_t)D_MODEL * D_MODEL * 2);
  bf16* Qb      = (bf16*)carve(NE * 2);
  bf16* Kb      = (bf16*)carve(NE * 2);
  bf16* Vt      = (bf16*)carve(NE * 2);
  bf16* Yb      = (bf16*)carve(NE * 2);

  int n1 = M_ROWS * D_MODEL;            // 8388608
  k_f32_to_bf16<<<n1 / 256, 256, 0, stream>>>(x, xb, n1);

  int n2 = D_MODEL * 3 * D_MODEL;       // 3145728
  k_pack_w<<<n2 / 256, 256, 0, stream>>>(wqkv, wqkv_p, 3 * D_MODEL, (3 * D_MODEL) / 16, n2);

  int n3 = D_MODEL * D_MODEL;           // 1048576
  k_pack_w<<<n3 / 256, 256, 0, stream>>>(wproj, wproj_p, D_MODEL, D_MODEL / 16, n3);

  // QKV GEMM: (8192/32) row tiles * (3072/64) col groups = 12288 waves / 8 per block
  k_gemm_qkv<<<(256 * 48) / 8, 256, 0, stream>>>(xb, wqkv_p, Qb, Kb, Vt);

  // Attention: 64 (b,h) * 128 q-tiles = 8192 waves / 8 per block
  k_attention<<<(64 * 128) / 8, 256, 0, stream>>>(Qb, Kb, Vt, Yb);

  // Projection: (8192/32) row tiles * (1024/64) col groups = 4096 waves / 8 per block
  k_gemm_proj<<<(256 * 16) / 8, 256, 0, stream>>>(Yb, wproj_p, out);
}
